// EncoderLayer_83958020702478
// MI455X (gfx1250) — matrix operand order repaired and checked by reading, not compiled, not run on hardware
//
#include <hip/hip_runtime.h>
#include <hip/hip_bf16.h>

typedef __attribute__((ext_vector_type(16))) __bf16        v16bf;
typedef __attribute__((ext_vector_type(8)))  float         v8f;
typedef __attribute__((ext_vector_type(8)))  unsigned int  v8u;

#define NT   8192   // tokens = B*S
#define DMOD 1024
#define DFF  4096
#define SLEN 2048
#define NH   16

__device__ __forceinline__ unsigned short f2bf(float f) {
    unsigned u = __float_as_uint(f);
    unsigned r = 0x7FFFu + ((u >> 16) & 1u);
    return (unsigned short)((u + r) >> 16);
}

// CDNA5 async global->LDS copy (ASYNCcnt-tracked, no VGPR staging).
// lds_off = wave-relative DS byte address (low 32 bits of the generic pointer).
__device__ __forceinline__ void async_copy_b128(const unsigned short* lds_ptr,
                                                const unsigned short* gptr) {
    unsigned lds_off = (unsigned)(unsigned long long)lds_ptr;
    asm volatile("global_load_async_to_lds_b128 %0, %1, off"
                 :: "v"(lds_off), "v"(gptr)
                 : "memory");
}
__device__ __forceinline__ void wait_async0() {
    asm volatile("s_wait_asynccnt 0" ::: "memory");
}

// ---------------------------------------------------------------- convert (row-major)
__global__ __launch_bounds__(256) void cvt_f32_bf16(const float* __restrict__ s,
                                                    unsigned short* __restrict__ d,
                                                    int n) {
    int i = blockIdx.x * 256 + threadIdx.x;
    if (i < n) d[i] = f2bf(s[i]);
}

// ---------------------------------------------------------------- convert + transpose
// in: f32 [R][C]  ->  out: bf16 [C][R].  32x32 LDS tile, coalesced both sides.
__global__ __launch_bounds__(256) void transpose_cvt_kernel(const float* __restrict__ in,
                                                            unsigned short* __restrict__ out,
                                                            int R, int C) {
    __shared__ unsigned short tile[32][33];
    const int bx = blockIdx.x * 32;           // col base (input)
    const int by = blockIdx.y * 32;           // row base (input)
    const int tx = threadIdx.x & 31;
    const int ty = threadIdx.x >> 5;          // 0..7
    for (int i = 0; i < 4; i++) {
        int r = ty + i * 8;
        tile[r][tx] = f2bf(in[(long)(by + r) * C + bx + tx]);
    }
    __syncthreads();
    for (int i = 0; i < 4; i++) {
        int c = ty + i * 8;                   // output row = original col
        out[(long)(bx + c) * R + by + tx] = tile[tx][c];
    }
}

// ---------------------------------------------------------------- GEMM
// C[M,N] = A[M,K](bf16, row-major) x Bt[N,K](bf16, K-major = B transposed), f32 accum.
// Epilogue: +residual(f32, optional), ReLU(optional), store f32 and/or bf16.
// WG = 256 thr = 8 waves; tile 128x128; wave tile 64x32; K-step 32;
// double-buffered LDS filled with global_load_async_to_lds_b128.
__global__ __launch_bounds__(256) void gemm_bf16_kernel(
    const unsigned short* __restrict__ A,
    const unsigned short* __restrict__ Bt,
    const float* __restrict__ residual,
    float* __restrict__ outF,
    unsigned short* __restrict__ outB,
    int M, int N, int K, int relu)
{
    __shared__ unsigned short sA[2][128 * 32];   // [m][k]
    __shared__ unsigned short sB[2][128 * 32];   // [n][k]

    const int tid  = threadIdx.x;
    const int lane = tid & 31;
    const int wave = tid >> 5;
    const int wm   = wave >> 2;            // 0..1 (64 rows each)
    const int wn   = wave & 3;             // 0..3 (32 cols each)
    const int hl   = lane >> 4;            // lane half
    const int l16  = lane & 15;
    const long bm  = (long)blockIdx.y * 128;
    const long bn  = (long)blockIdx.x * 128;
    const int  r   = tid >> 2;             // 0..63
    const int  c   = (tid & 3) * 8;        // 0,8,16,24

    v8f acc[4][2];
    for (int i = 0; i < 4; i++)
        for (int j = 0; j < 2; j++)
            for (int e = 0; e < 8; e++) acc[i][j][e] = 0.0f;

    auto fill = [&](int buf, int k0) {
        async_copy_b128(&sA[buf][r * 32 + c],        A  + (bm + r)      * (long)K + k0 + c);
        async_copy_b128(&sA[buf][(r + 64) * 32 + c], A  + (bm + r + 64) * (long)K + k0 + c);
        async_copy_b128(&sB[buf][r * 32 + c],        Bt + (bn + r)      * (long)K + k0 + c);
        async_copy_b128(&sB[buf][(r + 64) * 32 + c], Bt + (bn + r + 64) * (long)K + k0 + c);
    };

    const int KT = K >> 5;
    fill(0, 0);

    for (int kt = 0; kt < KT; ++kt) {
        const int cur = kt & 1;
        wait_async0();          // my fill(cur) landed in LDS
        __syncthreads();        // everyone's fill(cur) landed; buf cur^1 free
        if (kt + 1 < KT) fill(cur ^ 1, (kt + 1) * 32);   // overlaps WMMA below

        // B fragments: lane = column n, vgpr v holds k-pair ((v<4?0:16)+hl*8+(v&3)*2, +1); two
        // 8-bf16 contiguous runs per fragment -> mergeable ds_load_b128 pairs.
        v16bf bfrag[2];
        for (int j = 0; j < 2; j++) {
            v8u t;
            int col = wn * 32 + j * 16 + l16;
            for (int v = 0; v < 8; v++) {
                t[v] = *reinterpret_cast<const unsigned int*>(
                    &sB[cur][col * 32 + ((v < 4) ? 0 : 16) + hl * 8 + (v & 3) * 2]);
            }
            bfrag[j] = __builtin_bit_cast(v16bf, t);
        }
        // A fragments: lane = row m, k = (v<4?0:16) + hl*8 + (v&3)*2
        for (int i = 0; i < 4; i++) {
            v8u t;
            int row = wm * 64 + i * 16 + l16;
            for (int v = 0; v < 8; v++) {
                int kk = ((v < 4) ? 0 : 16) + hl * 8 + (v & 3) * 2;
                t[v] = *reinterpret_cast<const unsigned int*>(&sA[cur][row * 32 + kk]);
            }
            v16bf afrag = __builtin_bit_cast(v16bf, t);
            for (int j = 0; j < 2; j++) {
                acc[i][j] = __builtin_amdgcn_wmma_f32_16x16x32_bf16(
                    false, afrag, false, bfrag[j], (short)0, acc[i][j], false, false);
            }
        }
    }

    // epilogue: C row = v + 8*hl, col = l16 within each 16x16 tile
    for (int i = 0; i < 4; i++) {
        for (int j = 0; j < 2; j++) {
            long gc = bn + wn * 32 + j * 16 + l16;
            for (int v = 0; v < 8; v++) {
                long gr = bm + wm * 64 + i * 16 + v + 8 * hl;
                float val = acc[i][j][v];
                if (residual) val += residual[gr * N + gc];
                if (relu)     val = fmaxf(val, 0.0f);
                if (outF)     outF[gr * N + gc] = val;
                if (outB)     outB[gr * N + gc] = f2bf(val);
            }
        }
    }
}

// ---------------------------------------------------------------- flash attention
// Q,K,V,O: [NT, DMOD] bf16, head h at cols h*64..h*64+63. One wave = 16 q rows,
// 8 waves share one (b,h); double-buffered 32-key K/V tiles via async LDS fills.
__global__ __launch_bounds__(256) void flash_attn_kernel(
    const unsigned short* __restrict__ Q,
    const unsigned short* __restrict__ Km,
    const unsigned short* __restrict__ V,
    unsigned short* __restrict__ O)
{
    __shared__ unsigned short sP[8][16 * 32];   // per-wave P staging (C->A transpose)
    __shared__ unsigned short sK[2][32 * 64];   // [key][d]
    __shared__ unsigned short sV[2][32 * 64];   // [key][d]

    const int tid  = threadIdx.x;
    const int lane = tid & 31;
    const int wave = tid >> 5;
    const int hl   = lane >> 4;
    const int l16  = lane & 15;
    const int h    = blockIdx.y;
    const int b    = blockIdx.z;
    const int q0   = blockIdx.x * 128 + wave * 16;
    const long base = ((long)b * SLEN) * DMOD + h * 64;
    const float scale = 0.125f;   // 1/sqrt(64)
    const int fk = tid >> 3;             // 0..31 : key row for the cooperative fill
    const int fd = (tid & 7) * 8;        // 0,8,..,56 : d column

    auto fill = [&](int buf, int kk0) {
        async_copy_b128(&sK[buf][fk * 64 + fd], Km + base + (long)(kk0 + fk) * DMOD + fd);
        async_copy_b128(&sV[buf][fk * 64 + fd], V  + base + (long)(kk0 + fk) * DMOD + fd);
    };

    // Q as two A fragments (d = 0..31, 32..63)
    v16bf aQ[2];
    for (int ko = 0; ko < 2; ko++) {
        v8u t;
        for (int v = 0; v < 8; v++) {
            int kk = ko * 32 + ((v < 4) ? 0 : 16) + hl * 8 + (v & 3) * 2;
            t[v] = *reinterpret_cast<const unsigned int*>(
                Q + base + (long)(q0 + l16) * DMOD + kk);
        }
        aQ[ko] = __builtin_bit_cast(v16bf, t);
    }

    float rm[8], rl[8];
    v8f accO[4];
    for (int v = 0; v < 8; v++) { rm[v] = -3.0e38f; rl[v] = 0.0f; }
    for (int j = 0; j < 4; j++)
        for (int e = 0; e < 8; e++) accO[j][e] = 0.0f;

    const int ITER = SLEN / 32;
    fill(0, 0);

    for (int it = 0; it < ITER; ++it) {
        const int cur = it & 1;
        wait_async0();
        __syncthreads();
        if (it + 1 < ITER) fill(cur ^ 1, (it + 1) * 32);   // overlaps compute

        // scores: two 16x16 tiles, each Q(16x64) x K^T(64x16) = 2 WMMAs
        v8f st[2];
        for (int t2 = 0; t2 < 2; t2++) {
            for (int e = 0; e < 8; e++) st[t2][e] = 0.0f;
            int key = t2 * 16 + l16;
            for (int ko = 0; ko < 2; ko++) {
                v8u t;
                for (int v = 0; v < 8; v++) {
                    t[v] = *reinterpret_cast<const unsigned int*>(
                        &sK[cur][key * 64 + ko * 32 + ((v < 4) ? 0 : 16) + hl * 8 + (v & 3) * 2]);
                }
                v16bf bK = __builtin_bit_cast(v16bf, t);
                st[t2] = __builtin_amdgcn_wmma_f32_16x16x32_bf16(
                    false, aQ[ko], false, bK, (short)0, st[t2], false, false);
            }
        }
        // online softmax per row (row = v + 8*hl, owned by one 16-lane half)
        for (int v = 0; v < 8; v++) {
            float s0 = st[0][v] * scale;
            float s1 = st[1][v] * scale;
            float tm = fmaxf(s0, s1);
            for (int m2 = 1; m2 < 16; m2 <<= 1) tm = fmaxf(tm, __shfl_xor(tm, m2, 32));
            float nm   = fmaxf(rm[v], tm);
            float corr = __expf(rm[v] - nm);
            rm[v] = nm;
            float p0 = __expf(s0 - nm);
            float p1 = __expf(s1 - nm);
            float rs = p0 + p1;
            for (int m2 = 1; m2 < 16; m2 <<= 1) rs += __shfl_xor(rs, m2, 32);
            rl[v] = rl[v] * corr + rs;
            for (int j = 0; j < 4; j++) accO[j][v] *= corr;
            int row = v + 8 * hl;
            sP[wave][row * 32 + l16]      = f2bf(p0);
            sP[wave][row * 32 + 16 + l16] = f2bf(p1);
        }
        // P (16x32) back as A fragment (same-wave DS is in-order)
        v8u tp;
        for (int v = 0; v < 8; v++) {
            int kk = ((v < 4) ? 0 : 16) + hl * 8 + (v & 3) * 2;
            tp[v] = *reinterpret_cast<const unsigned int*>(&sP[wave][l16 * 32 + kk]);
        }
        v16bf aP = __builtin_bit_cast(v16bf, tp);
        // context += P(16x32) x V(32x64): 4 d-tiles
        for (int j = 0; j < 4; j++) {
            v8u tv;
            int d = j * 16 + l16;
            for (int v = 0; v < 8; v++) {
                int key = ((v < 4) ? 0 : 16) + hl * 8 + (v & 3) * 2;
                unsigned lo = sV[cur][key * 64 + d];
                unsigned hi = sV[cur][(key + 1) * 64 + d];
                tv[v] = lo | (hi << 16);
            }
            v16bf bV = __builtin_bit_cast(v16bf, tv);
            accO[j] = __builtin_amdgcn_wmma_f32_16x16x32_bf16(
                false, aP, false, bV, (short)0, accO[j], false, false);
        }
    }

    // finalize: divide by row sums, store bf16 context
    for (int v = 0; v < 8; v++) {
        float inv = 1.0f / rl[v];
        long row = (long)b * SLEN + q0 + v + 8 * hl;
        for (int j = 0; j < 4; j++) {
            int col = h * 64 + j * 16 + l16;
            O[row * DMOD + col] = f2bf(accO[j][v] * inv);
        }
    }
}

// ---------------------------------------------------------------- LayerNorm
// one block per row of 1024; population variance (jnp.var, ddof=0); in-place safe.
__global__ __launch_bounds__(256) void layernorm_kernel(
    const float* __restrict__ x, const float* __restrict__ g, const float* __restrict__ bb,
    float* __restrict__ outF, unsigned short* __restrict__ outB)
{
    __shared__ float red[256];
    const int tid = threadIdx.x;
    const long row = blockIdx.x;
    const float* xr = x + row * DMOD;

    float v[4];
    float s = 0.0f;
    for (int i = 0; i < 4; i++) { v[i] = xr[tid + i * 256]; s += v[i]; }
    red[tid] = s; __syncthreads();
    for (int st = 128; st > 0; st >>= 1) { if (tid < st) red[tid] += red[tid + st]; __syncthreads(); }
    float mean = red[0] * (1.0f / DMOD);
    __syncthreads();

    float sq = 0.0f;
    for (int i = 0; i < 4; i++) { float d = v[i] - mean; sq += d * d; }
    red[tid] = sq; __syncthreads();
    for (int st = 128; st > 0; st >>= 1) { if (tid < st) red[tid] += red[tid + st]; __syncthreads(); }
    float var  = red[0] * (1.0f / DMOD);
    float rstd = rsqrtf(var + 1e-5f);

    for (int i = 0; i < 4; i++) {
        int c = tid + i * 256;
        float y = (v[i] - mean) * rstd * g[c] + bb[c];
        if (outF) outF[row * DMOD + c] = y;
        if (outB) outB[row * DMOD + c] = f2bf(y);
    }
}

// ---------------------------------------------------------------- launch
extern "C" void kernel_launch(void* const* d_in, const int* in_sizes, int n_in,
                              void* d_out, int out_size, void* d_ws, size_t ws_size,
                              hipStream_t stream) {
    (void)in_sizes; (void)n_in; (void)out_size; (void)ws_size;
    const float* X    = (const float*)d_in[0];
    const float* Wq   = (const float*)d_in[1];
    const float* Wk   = (const float*)d_in[2];
    const float* Wv   = (const float*)d_in[3];
    const float* Wo   = (const float*)d_in[4];
    const float* W1   = (const float*)d_in[5];
    const float* W2   = (const float*)d_in[6];
    const float* ln1g = (const float*)d_in[7];
    const float* ln1b = (const float*)d_in[8];
    const float* ln2g = (const float*)d_in[9];
    const float* ln2b = (const float*)d_in[10];
    float* Out = (float*)d_out;

    char* ws = (char*)d_ws;
    size_t off = 0;
    auto alloc = [&](size_t bytes) -> char* {
        char* p = ws + off;
        off += (bytes + 255) & ~(size_t)255;
        return p;
    };
    unsigned short* Xb    = (unsigned short*)alloc((size_t)NT * DMOD * 2);
    unsigned short* Wqb   = (unsigned short*)alloc((size_t)DMOD * DMOD * 2);   // [N][K]
    unsigned short* Wkb   = (unsigned short*)alloc((size_t)DMOD * DMOD * 2);
    unsigned short* Wvb   = (unsigned short*)alloc((size_t)DMOD * DMOD * 2);
    unsigned short* Wob   = (unsigned short*)alloc((size_t)DMOD * DMOD * 2);
    unsigned short* W1b   = (unsigned short*)alloc((size_t)DMOD * DFF * 2);    // [DFF][DMOD]
    unsigned short* W2b   = (unsigned short*)alloc((size_t)DFF * DMOD * 2);    // [DMOD][DFF]
    unsigned short* Qb    = (unsigned short*)alloc((size_t)NT * DMOD * 2);
    unsigned short* Kb    = (unsigned short*)alloc((size_t)NT * DMOD * 2);
    unsigned short* Vb    = (unsigned short*)alloc((size_t)NT * DMOD * 2);
    unsigned short* Cb    = (unsigned short*)alloc((size_t)NT * DMOD * 2);
    float*          AttnF = (float*)alloc((size_t)NT * DMOD * 4);
    unsigned short* AttnB = (unsigned short*)alloc((size_t)NT * DMOD * 2);
    // H (NT x DFF bf16, 64 MiB) reuses the contiguous Q/K/V/Ctx region (4 x 16 MiB),
    // dead by the time FFN1 runs.
    unsigned short* Hb = Qb;

    dim3 blk(256);
    // activations: row-major bf16
    cvt_f32_bf16<<<dim3((unsigned)(((long)NT * DMOD + 255) / 256)), blk, 0, stream>>>(
        X, Xb, NT * DMOD);
    // weights: transposed to [N][K] bf16
    transpose_cvt_kernel<<<dim3(DMOD / 32, DMOD / 32), blk, 0, stream>>>(Wq, Wqb, DMOD, DMOD);
    transpose_cvt_kernel<<<dim3(DMOD / 32, DMOD / 32), blk, 0, stream>>>(Wk, Wkb, DMOD, DMOD);
    transpose_cvt_kernel<<<dim3(DMOD / 32, DMOD / 32), blk, 0, stream>>>(Wv, Wvb, DMOD, DMOD);
    transpose_cvt_kernel<<<dim3(DMOD / 32, DMOD / 32), blk, 0, stream>>>(Wo, Wob, DMOD, DMOD);
    transpose_cvt_kernel<<<dim3(DFF / 32,  DMOD / 32), blk, 0, stream>>>(W1, W1b, DMOD, DFF);
    transpose_cvt_kernel<<<dim3(DMOD / 32, DFF / 32),  blk, 0, stream>>>(W2, W2b, DFF, DMOD);

    // Q/K/V projections
    gemm_bf16_kernel<<<dim3(8, 64), blk, 0, stream>>>(Xb, Wqb, nullptr, nullptr, Qb, NT, DMOD, DMOD, 0);
    gemm_bf16_kernel<<<dim3(8, 64), blk, 0, stream>>>(Xb, Wkb, nullptr, nullptr, Kb, NT, DMOD, DMOD, 0);
    gemm_bf16_kernel<<<dim3(8, 64), blk, 0, stream>>>(Xb, Wvb, nullptr, nullptr, Vb, NT, DMOD, DMOD, 0);

    // attention -> context (bf16)
    flash_attn_kernel<<<dim3(SLEN / 128, NH, 4), blk, 0, stream>>>(Qb, Kb, Vb, Cb);

    // O projection + residual (f32 into d_out), then LN1 -> AttnF/AttnB
    gemm_bf16_kernel<<<dim3(8, 64), blk, 0, stream>>>(Cb, Wob, X, Out, nullptr, NT, DMOD, DMOD, 0);
    layernorm_kernel<<<dim3(NT), blk, 0, stream>>>(Out, ln1g, ln1b, AttnF, AttnB);

    // FFN: relu(AttnB @ W1) -> Hb;  Hb @ W2 + AttnF -> d_out;  LN2 in-place
    gemm_bf16_kernel<<<dim3(32, 64), blk, 0, stream>>>(AttnB, W1b, nullptr, nullptr, Hb, NT, DFF, DMOD, 1);
    gemm_bf16_kernel<<<dim3(8, 64), blk, 0, stream>>>(Hb, W2b, AttnF, Out, nullptr, NT, DMOD, DFF, 0);
    layernorm_kernel<<<dim3(NT), blk, 0, stream>>>(Out, ln2g, ln2b, Out, nullptr);
}
